// HydraMambaSegV1_34265249088355
// MI455X (gfx1250) — matrix operand order repaired and checked by reading, not compiled, not run on hardware
//
#include <hip/hip_runtime.h>
#include <hip/hip_bf16.h>

// ---------------- problem constants ----------------
constexpr int CB   = 4;        // batch
constexpr int CL   = 2048;     // sequence
constexpr int CC   = 384;      // channels
constexpr int CH   = 12;       // heads
constexpr int CDH  = 64;       // head dim
constexpr int CN   = 64;       // state dim
constexpr int CDI  = 768;      // expand*C
constexpr int CPROJ= 1676;     // 2*DI + 2*N + H
constexpr int CPP  = 1680;     // PROJ padded to /16 (token stride 3360 B, 16B aligned)
constexpr int CHID = 1536;     // MLP hidden
constexpr int CM   = CB * CL;  // 8192 tokens
constexpr int CQ   = 64;       // SSD chunk length
constexpr int NCH  = CL / CQ;  // 32 chunks
constexpr int LDSS = 72;       // padded LDS row stride (f16), 144B: 16B aligned, conflict-free rows
constexpr float CINVS = 0.9999950000374997f; // 1/sqrt(1+1e-5)

typedef __attribute__((ext_vector_type(16))) _Float16 v16h;
typedef __attribute__((ext_vector_type(8)))  _Float16 v8h;
typedef __attribute__((ext_vector_type(8)))  float    v8f;

// async global->LDS b128 copy (CDNA5 GLOBAL_LOAD_ASYNC_TO_LDS_B128, ASYNCcnt-tracked).
// ldsOff = LDS byte address (low 32 bits of generic shared pointer), gaddr = 64-bit global address.
__device__ inline void async_g2l_b128(unsigned ldsOff, unsigned long long gaddr) {
    asm volatile("global_load_async_to_lds_b128 %0, %1, off"
                 :: "v"(ldsOff), "v"(gaddr) : "memory");
}
__device__ inline void wait_asynccnt0() {
    asm volatile("s_wait_asynccnt 0x0" ::: "memory");
}
__device__ inline unsigned lds_addr(const void* p) {
    return (unsigned)(unsigned long long)(size_t)p;   // LDS aperture: low 32 bits = LDS offset
}

// ---------------- WMMA fragment builders (CDNA5 ISA 7.12.2 lane maps) ----------------
template <typename F>
__device__ inline v16h build_a(int lane, F&& f) {
    int m = lane & 15, hi = lane >> 4;
    v16h r;
#pragma unroll
    for (int e = 0; e < 8; e++) {
        r[e]     = f(m, hi * 8 + e);
        r[e + 8] = f(m, 16 + hi * 8 + e);
    }
    return r;
}
template <typename F>
__device__ inline v16h build_b(int lane, F&& f) {
    int n = lane & 15, hi = lane >> 4;
    v16h r;
#pragma unroll
    for (int e = 0; e < 8; e++) { r[e] = f(n, hi * 8 + e); r[e + 8] = f(n, 16 + hi * 8 + e); }
    return r;
}

// ---------------- weight prep ----------------
__global__ void transpose_cvt_kernel(const float* __restrict__ src, _Float16* __restrict__ dst,
                                     int K, int Ncols, int Npad) {
    int idx = blockIdx.x * 256 + threadIdx.x;
    if (idx >= Npad * K) return;
    int n = idx / K, k = idx - n * K;
    dst[idx] = (_Float16)((n < Ncols) ? src[(size_t)k * Ncols + n] : 0.f);
}

__global__ void copy_cvt_kernel(const float* __restrict__ src, _Float16* __restrict__ dst, int total) {
    int idx = blockIdx.x * 256 + threadIdx.x;
    if (idx < total) dst[idx] = (_Float16)src[idx];
}

__global__ void pad_bias_kernel(const float* __restrict__ src, float* __restrict__ dst, int n, int npad) {
    int idx = blockIdx.x * 256 + threadIdx.x;
    if (idx < npad) dst[idx] = (idx < n) ? src[idx] : 0.f;
}

// ---------------- layernorm (f32 in -> f16 out) ----------------
__global__ void ln_kernel(const float* __restrict__ x, const float* __restrict__ g,
                          const float* __restrict__ be, _Float16* __restrict__ o) {
    int row = blockIdx.x;
    int t = threadIdx.x;
    const float* xr = x + (size_t)row * CC;
    float v[3], s = 0.f;
#pragma unroll
    for (int i = 0; i < 3; i++) { v[i] = xr[t + i * 128]; s += v[i]; }
    __shared__ float red[4];
#pragma unroll
    for (int off = 16; off; off >>= 1) s += __shfl_xor(s, off, 32);
    if ((t & 31) == 0) red[t >> 5] = s;
    __syncthreads();
    float mu = (red[0] + red[1] + red[2] + red[3]) * (1.f / CC);
    float q = 0.f;
#pragma unroll
    for (int i = 0; i < 3; i++) { float d = v[i] - mu; q += d * d; }
#pragma unroll
    for (int off = 16; off; off >>= 1) q += __shfl_xor(q, off, 32);
    __syncthreads();
    if ((t & 31) == 0) red[t >> 5] = q;
    __syncthreads();
    float rs = rsqrtf((red[0] + red[1] + red[2] + red[3]) * (1.f / CC) + 1e-5f);
    _Float16* orow = o + (size_t)row * CC;
#pragma unroll
    for (int i = 0; i < 3; i++) {
        int c = t + i * 128;
        orow[c] = (_Float16)((v[i] - mu) * rs * g[c] + be[c]);
    }
}

// ---------------- generic WMMA GEMM with fused epilogues ----------------
__device__ inline v8f wmma_tile(const _Float16* __restrict__ A, int lda,
                                const _Float16* __restrict__ Bt, int ldb,
                                int m0, int n0, int K, int lane, v8f acc) {
    int row = lane & 15, hi = lane >> 4;
    const _Float16* ap = A + (size_t)(m0 + row) * lda + hi * 16;
    const _Float16* bp = Bt + (size_t)(n0 + row) * ldb + hi * 16;
    for (int k0 = 0; k0 < K; k0 += 32) {
        v8h alo = *(const v8h*)(ap);
        v8h ahi = *(const v8h*)(ap + 8);
        v16h bf = *(const v16h*)(bp);
        v16h af;
#pragma unroll
        for (int i = 0; i < 8; i++) { af[i] = alo[i]; af[8 + i] = ahi[i]; }
        acc = __builtin_amdgcn_wmma_f32_16x16x32_f16(false, af, false, bf,
                                                     (short)0, acc, false, false);
        ap += 32; bp += 32;
    }
    return acc;
}

// mode 0: outH = f16(acc + bias[n])
// mode 1: outH = f16(relu((acc + bias[n]) * scale[n]*CINVS + shift[n]))
// mode 2: outF = acc1 + acc2 + bias[n] + bias2[n] + resid[m][n]
// mode 3: outF = acc + bias[n] + resid[m][n]
__global__ void gemm_kernel(const _Float16* __restrict__ A, int lda,
                            const _Float16* __restrict__ Bt, int ldb, int K,
                            const _Float16* __restrict__ A2, int lda2,
                            const _Float16* __restrict__ B2t, int ldb2, int K2,
                            int M, int Npad,
                            const float* __restrict__ bias, const float* __restrict__ bias2,
                            const float* __restrict__ scale, const float* __restrict__ shift,
                            const float* __restrict__ resid, int ldr,
                            _Float16* __restrict__ outH, float* __restrict__ outF, int ldo,
                            int mode) {
    int w = threadIdx.x >> 5;
    int lane = threadIdx.x & 31;
    int m0 = blockIdx.x * 32 + (w >> 2) * 16;
    int n0 = blockIdx.y * 64 + (w & 3) * 16;
    if (m0 >= M || n0 >= Npad) return;   // wave-uniform

    v8f acc = {};
    acc = wmma_tile(A, lda, Bt, ldb, m0, n0, K, lane, acc);
    if (mode == 2)
        acc = wmma_tile(A2, lda2, B2t, ldb2, m0, n0, K2, lane, acc);

    int row = lane & 15, hi = lane >> 4;
    int n = n0 + row;
    float bn  = bias  ? bias[n]  : 0.f;
    float b2n = (mode == 2 && bias2) ? bias2[n] : 0.f;
    float sc = 0.f, sh = 0.f;
    if (mode == 1) { sc = scale[n] * CINVS; sh = shift[n]; }

#pragma unroll
    for (int r = 0; r < 8; r++) {
        int m = m0 + hi * 8 + r;
        size_t o = (size_t)m * ldo + n;
        float v = acc[r] + bn;
        if (mode == 0) {
            outH[o] = (_Float16)v;
        } else if (mode == 1) {
            v = v * sc + sh;
            outH[o] = (_Float16)(v > 0.f ? v : 0.f);
        } else if (mode == 2) {
            outF[o] = v + b2n + resid[(size_t)m * ldr + n];
        } else {
            outF[o] = v + resid[(size_t)m * ldr + n];
        }
    }
}

// ================= chunked SSD scan =================
// Pass 1: per (b,h,chunk): T[d][n] = sum_j X[j][d]*e^{Ptot-P_j}*dt_j*B[j][n] (WMMA), Etot = e^{Ptot}.
__global__ void __launch_bounds__(256)
chunk_state_kernel(const _Float16* __restrict__ Z,
                   const float* __restrict__ dtb, const float* __restrict__ Alog,
                   _Float16* __restrict__ Sbuf, float* __restrict__ Etot, int rev) {
    int c = blockIdx.x % NCH;
    int h = (blockIdx.x / NCH) % CH;
    int b = blockIdx.x / (NCH * CH);
    int t0 = c * CQ;
    int tid = threadIdx.x;
    float A = -__expf(Alog[h]);
    float dtbh = dtb[h];

    __shared__ __attribute__((aligned(16))) _Float16 sBt[CQ * LDSS];   // [n][j]
    __shared__ __attribute__((aligned(16))) _Float16 sXt[CQ * LDSS];   // [d][j]
    __shared__ float sdt[CQ], sP[CQ], sW[CQ];

    for (int u = tid >> 6; u < CQ; u += 4) {    // transposed staging (no async B16)
        int e = tid & 63;
        int l = rev ? (CL - 1 - (t0 + u)) : (t0 + u);
        const _Float16* zp = Z + ((size_t)b * CL + l) * CPP;
        sBt[e * LDSS + u] = zp[2 * CDI + e];
        sXt[e * LDSS + u] = zp[CDI + h * CDH + e];
    }
    if (tid < CQ) {
        int l = rev ? (CL - 1 - (t0 + tid)) : (t0 + tid);
        float x = (float)Z[((size_t)b * CL + l) * CPP + 2 * CDI + 2 * CN + h] + dtbh;
        sdt[tid] = (x > 20.f) ? x : log1pf(__expf(x));
    }
    __syncthreads();
    if (tid == 0) {
        float run = 0.f;
        for (int u = 0; u < CQ; u++) { run += sdt[u] * A; sP[u] = run; }
        Etot[blockIdx.x] = __expf(run);
    }
    __syncthreads();
    if (tid < CQ) sW[tid] = __expf(sP[CQ - 1] - sP[tid]) * sdt[tid];
    __syncthreads();

    int w = tid >> 5, lane = tid & 31;
    size_t base = (size_t)blockIdx.x * (CQ * CN);
    for (int tile = w; tile < 16; tile += 8) {
        int tm = (tile >> 2) * 16, tn = (tile & 3) * 16;   // tm: d, tn: n
        v8f acc = {};
        for (int k0 = 0; k0 < CQ; k0 += 32) {
            v16h a = build_a(lane, [&](int m, int k) {
                return (_Float16)((float)sXt[(tm + m) * LDSS + k0 + k] * sW[k0 + k]);
            });
            v16h bf = build_b(lane, [&](int n, int k) {
                return sBt[(tn + n) * LDSS + k0 + k];
            });
            acc = __builtin_amdgcn_wmma_f32_16x16x32_f16(false, a, false, bf,
                                                         (short)0, acc, false, false);
        }
        int col = lane & 15, hi = lane >> 4;
#pragma unroll
        for (int r = 0; r < 8; r++) {
            int d = tm + hi * 8 + r;
            Sbuf[base + d * CN + tn + col] = (_Float16)acc[r];
        }
    }
}

// Pass 1.5: serial prefix over the 32 chunk states: slot c <- S_before_c.
__global__ void __launch_bounds__(256)
prefix_kernel(_Float16* __restrict__ Sbuf, const float* __restrict__ Etot) {
    int bh = blockIdx.x;
    int tid = threadIdx.x;
    float S[16];
#pragma unroll
    for (int i = 0; i < 16; i++) S[i] = 0.f;
    for (int c = 0; c < NCH; c++) {
        size_t base = ((size_t)bh * NCH + c) * (CQ * CN);
        float e = Etot[bh * NCH + c];
#pragma unroll
        for (int i = 0; i < 16; i++) {
            size_t idx = base + (size_t)tid * 16 + i;
            float t = (float)Sbuf[idx];
            Sbuf[idx] = (_Float16)S[i];
            S[i] = e * S[i] + t;
        }
    }
}

// Pass 2: G = C@B^T (WMMA), causal decay mask, Y = Gm@X + (C*e^{P_i})@S_prev (WMMA),
// fused D-skip + SiLU gate. Row-contiguous operands staged via async global->LDS b128.
__global__ void __launch_bounds__(256)
chunk_out_kernel(const _Float16* __restrict__ Z,
                 const float* __restrict__ dtb, const float* __restrict__ Alog,
                 const float* __restrict__ Dv,
                 const _Float16* __restrict__ Sbuf,
                 _Float16* __restrict__ ys, int rev) {
    int c = blockIdx.x % NCH;
    int h = (blockIdx.x / NCH) % CH;
    int b = blockIdx.x / (NCH * CH);
    int t0 = c * CQ;
    int tid = threadIdx.x;
    float A = -__expf(Alog[h]);
    float dtbh = dtb[h];

    __shared__ __attribute__((aligned(16))) _Float16 sC[CQ * LDSS];    // [i][n]
    __shared__ __attribute__((aligned(16))) _Float16 sB[CQ * LDSS];    // [j][n]
    __shared__ __attribute__((aligned(16))) _Float16 sXt[CQ * LDSS];   // [d][j]
    __shared__ __attribute__((aligned(16))) _Float16 sZg[CQ * LDSS];   // [i][d]
    __shared__ __attribute__((aligned(16))) _Float16 sG[CQ * LDSS];    // [i][j] masked
    __shared__ __attribute__((aligned(16))) _Float16 sSt[CQ * LDSS];   // [d][n]
    __shared__ float sdt[CQ], sP[CQ], sEP[CQ];

    // ---- async staging of row-contiguous tiles: each thread issues 4x b128 ----
    {
        int u = tid >> 2, q = tid & 3;       // token row, 16-half segment
        int l = rev ? (CL - 1 - (t0 + u)) : (t0 + u);
        const _Float16* zp = Z + ((size_t)b * CL + l) * CPP;
        async_g2l_b128(lds_addr(&sB[u * LDSS + q * 16]),
                       (unsigned long long)(size_t)(zp + 2 * CDI + q * 16));
        async_g2l_b128(lds_addr(&sC[u * LDSS + q * 16]),
                       (unsigned long long)(size_t)(zp + 2 * CDI + CN + q * 16));
        async_g2l_b128(lds_addr(&sZg[u * LDSS + q * 16]),
                       (unsigned long long)(size_t)(zp + h * CDH + q * 16));
        const _Float16* sp = Sbuf + (size_t)blockIdx.x * (CQ * CN) + u * CN + q * 16;
        async_g2l_b128(lds_addr(&sSt[u * LDSS + q * 16]),
                       (unsigned long long)(size_t)sp);
    }
    // ---- transposed X staging (needs element transpose; keep VGPR path) ----
    for (int u = tid >> 6; u < CQ; u += 4) {
        int e = tid & 63;
        int l = rev ? (CL - 1 - (t0 + u)) : (t0 + u);
        sXt[e * LDSS + u] = Z[((size_t)b * CL + l) * CPP + CDI + h * CDH + e];
    }
    if (tid < CQ) {
        int l = rev ? (CL - 1 - (t0 + tid)) : (t0 + tid);
        float x = (float)Z[((size_t)b * CL + l) * CPP + 2 * CDI + 2 * CN + h] + dtbh;
        sdt[tid] = (x > 20.f) ? x : log1pf(__expf(x));
    }
    wait_asynccnt0();                         // async LDS writes landed (this wave)
    __syncthreads();                          // visible to all waves
    if (tid == 0) {
        float run = 0.f;
        for (int u = 0; u < CQ; u++) { run += sdt[u] * A; sP[u] = run; }
    }
    __syncthreads();
    if (tid < CQ) sEP[tid] = __expf(sP[tid]);
    __syncthreads();

    int w = tid >> 5, lane = tid & 31;

    // GEMM1: G[i][j] = C_i . B_j, causal mask * e^{P_i-P_j} * dt_j -> sG
    for (int tile = w; tile < 16; tile += 8) {
        int tm = (tile >> 2) * 16, tn = (tile & 3) * 16;   // tm: i, tn: j
        v8f acc = {};
        for (int k0 = 0; k0 < CN; k0 += 32) {
            v16h a = build_a(lane, [&](int m, int k) {
                return sC[(tm + m) * LDSS + k0 + k];
            });
            v16h bf = build_b(lane, [&](int n, int k) {
                return sB[(tn + n) * LDSS + k0 + k];
            });
            acc = __builtin_amdgcn_wmma_f32_16x16x32_f16(false, a, false, bf,
                                                         (short)0, acc, false, false);
        }
        int col = lane & 15, hi = lane >> 4;
        int j = tn + col;
        float Pj = sP[j], dtj = sdt[j];
#pragma unroll
        for (int r = 0; r < 8; r++) {
            int i = tm + hi * 8 + r;
            float g = (j <= i) ? acc[r] * __expf(sP[i] - Pj) * dtj : 0.f;
            sG[i * LDSS + j] = (_Float16)g;
        }
    }
    __syncthreads();

    // GEMM2 (+3): Y[i][d] = Gm@X + (C*e^{P_i})@S_prev, then D-skip + SiLU gate
    for (int tile = w; tile < 16; tile += 8) {
        int tm = (tile >> 2) * 16, tn = (tile & 3) * 16;   // tm: i, tn: d
        v8f acc = {};
        for (int k0 = 0; k0 < CQ; k0 += 32) {
            v16h a = build_a(lane, [&](int m, int k) {
                return sG[(tm + m) * LDSS + k0 + k];
            });
            v16h bf = build_b(lane, [&](int n, int k) {
                return sXt[(tn + n) * LDSS + k0 + k];
            });
            acc = __builtin_amdgcn_wmma_f32_16x16x32_f16(false, a, false, bf,
                                                         (short)0, acc, false, false);
        }
        for (int k0 = 0; k0 < CN; k0 += 32) {
            v16h a = build_a(lane, [&](int m, int k) {
                return (_Float16)((float)sC[(tm + m) * LDSS + k0 + k] * sEP[tm + m]);
            });
            v16h bf = build_b(lane, [&](int n, int k) {
                return sSt[(tn + n) * LDSS + k0 + k];
            });
            acc = __builtin_amdgcn_wmma_f32_16x16x32_f16(false, a, false, bf,
                                                         (short)0, acc, false, false);
        }
        int col = lane & 15, hi = lane >> 4;
        int d = tn + col;
        float Dd = Dv[h * CDH + d];
#pragma unroll
        for (int r = 0; r < 8; r++) {
            int i = tm + hi * 8 + r;
            float y = acc[r] + Dd * (float)sXt[d * LDSS + i];
            float zv = (float)sZg[i * LDSS + d];
            float gt = zv / (1.f + __expf(-zv));
            int l = rev ? (CL - 1 - (t0 + i)) : (t0 + i);
            ys[((size_t)b * CL + l) * CDI + h * CDH + d] = (_Float16)(y * gt);
        }
    }
}

// ---------------- host orchestration ----------------
extern "C" void kernel_launch(void* const* d_in, const int* in_sizes, int n_in,
                              void* d_out, int out_size, void* d_ws, size_t ws_size,
                              hipStream_t stream) {
    const float* points = (const float*)d_in[0];
    const float* g1     = (const float*)d_in[1];
    const float* be1    = (const float*)d_in[2];
    const float* g2     = (const float*)d_in[3];
    const float* be2    = (const float*)d_in[4];
    const float* Winf   = (const float*)d_in[5];
    const float* binf   = (const float*)d_in[6];
    const float* dtbf   = (const float*)d_in[7];
    const float* Alogf  = (const float*)d_in[8];
    const float* Df     = (const float*)d_in[9];
    const float* Woutf  = (const float*)d_in[10];
    const float* boutf  = (const float*)d_in[11];
    const float* Winb   = (const float*)d_in[12];
    const float* binb   = (const float*)d_in[13];
    const float* dtbb   = (const float*)d_in[14];
    const float* Alogb  = (const float*)d_in[15];
    const float* Db     = (const float*)d_in[16];
    const float* Woutb  = (const float*)d_in[17];
    const float* boutb  = (const float*)d_in[18];
    const float* W1     = (const float*)d_in[19];
    const float* b1m    = (const float*)d_in[20];
    const float* bng    = (const float*)d_in[21];
    const float* bnb    = (const float*)d_in[22];
    const float* W2     = (const float*)d_in[23];
    const float* b2m    = (const float*)d_in[24];
    float* out = (float*)d_out;

    char* base = (char*)d_ws;
    size_t off = 0;
    auto take = [&](size_t bytes) -> char* {
        char* p = base + off;
        off = (off + bytes + 255) & ~(size_t)255;
        return p;
    };
    _Float16* xnh  = (_Float16*)take((size_t)CM * CC * 2);
    _Float16* Zf   = (_Float16*)take((size_t)CM * CPP * 2);
    _Float16* Zb   = (_Float16*)take((size_t)CM * CPP * 2);
    _Float16* ysf  = (_Float16*)take((size_t)CM * CDI * 2);
    _Float16* ysb  = (_Float16*)take((size_t)CM * CDI * 2);
    float*    x1   = (float*)   take((size_t)CM * CC * 4);
    _Float16* hn   = (_Float16*)take((size_t)CM * CC * 2);
    _Float16* h1   = (_Float16*)take((size_t)CM * CHID * 2);
    _Float16* WfT  = (_Float16*)take((size_t)CPP * CC * 2);
    _Float16* WbT  = (_Float16*)take((size_t)CPP * CC * 2);
    _Float16* WofT = (_Float16*)take((size_t)CC * CDI * 2);
    _Float16* WobT = (_Float16*)take((size_t)CC * CDI * 2);
    _Float16* W1h  = (_Float16*)take((size_t)CHID * CC * 2);
    _Float16* W2h  = (_Float16*)take((size_t)CC * CHID * 2);
    float*    binfp= (float*)   take((size_t)CPP * 4);
    float*    binbp= (float*)   take((size_t)CPP * 4);
    const size_t nchunks = (size_t)CB * CH * NCH;          // 1536 per dir
    _Float16* Sf   = (_Float16*)take(nchunks * CQ * CN * 2);
    _Float16* Sb   = (_Float16*)take(nchunks * CQ * CN * 2);
    float*    Etf  = (float*)   take(nchunks * 4);
    float*    Etb  = (float*)   take(nchunks * 4);

    // ---- weight prep ----
    {
        int tot = CPP * CC;
        transpose_cvt_kernel<<<(tot + 255) / 256, 256, 0, stream>>>(Winf, WfT, CC, CPROJ, CPP);
        transpose_cvt_kernel<<<(tot + 255) / 256, 256, 0, stream>>>(Winb, WbT, CC, CPROJ, CPP);
    }
    {
        int tot = CC * CDI;
        transpose_cvt_kernel<<<(tot + 255) / 256, 256, 0, stream>>>(Woutf, WofT, CDI, CC, CC);
        transpose_cvt_kernel<<<(tot + 255) / 256, 256, 0, stream>>>(Woutb, WobT, CDI, CC, CC);
    }
    {
        int tot = CHID * CC;
        copy_cvt_kernel<<<(tot + 255) / 256, 256, 0, stream>>>(W1, W1h, tot);
        copy_cvt_kernel<<<(tot + 255) / 256, 256, 0, stream>>>(W2, W2h, tot);
    }
    pad_bias_kernel<<<(CPP + 255) / 256, 256, 0, stream>>>(binf, binfp, CPROJ, CPP);
    pad_bias_kernel<<<(CPP + 255) / 256, 256, 0, stream>>>(binb, binbp, CPROJ, CPP);

    // ---- LN1 ----
    ln_kernel<<<CM, 128, 0, stream>>>(points, g1, be1, xnh);

    // ---- in-projections (WMMA, mode 0) ----
    {
        dim3 grid(CM / 32, (CPP + 63) / 64);
        gemm_kernel<<<grid, 256, 0, stream>>>(xnh, CC, WfT, CC, CC,
                                              nullptr, 0, nullptr, 0, 0,
                                              CM, CPP, binfp, nullptr, nullptr, nullptr,
                                              nullptr, 0, Zf, nullptr, CPP, 0);
        gemm_kernel<<<grid, 256, 0, stream>>>(xnh, CC, WbT, CC, CC,
                                              nullptr, 0, nullptr, 0, 0,
                                              CM, CPP, binbp, nullptr, nullptr, nullptr,
                                              nullptr, 0, Zb, nullptr, CPP, 0);
    }

    // ---- chunked SSD scan (WMMA) ----
    chunk_state_kernel<<<(int)nchunks, 256, 0, stream>>>(Zf, dtbf, Alogf, Sf, Etf, 0);
    chunk_state_kernel<<<(int)nchunks, 256, 0, stream>>>(Zb, dtbb, Alogb, Sb, Etb, 1);
    prefix_kernel<<<CB * CH, 256, 0, stream>>>(Sf, Etf);
    prefix_kernel<<<CB * CH, 256, 0, stream>>>(Sb, Etb);
    chunk_out_kernel<<<(int)nchunks, 256, 0, stream>>>(Zf, dtbf, Alogf, Df, Sf, ysf, 0);
    chunk_out_kernel<<<(int)nchunks, 256, 0, stream>>>(Zb, dtbb, Alogb, Db, Sb, ysb, 1);

    // ---- out-projection (mode 2) ----
    {
        dim3 grid(CM / 32, CC / 64);
        gemm_kernel<<<grid, 256, 0, stream>>>(ysf, CDI, WofT, CDI, CDI,
                                              ysb, CDI, WobT, CDI, CDI,
                                              CM, CC, boutf, boutb, nullptr, nullptr,
                                              points, CC, nullptr, x1, CC, 2);
    }

    // ---- LN2 ----
    ln_kernel<<<CM, 128, 0, stream>>>(x1, g2, be2, hn);

    // ---- MLP1 (mode 1) ----
    {
        dim3 grid(CM / 32, CHID / 64);
        gemm_kernel<<<grid, 256, 0, stream>>>(hn, CC, W1h, CC, CC,
                                              nullptr, 0, nullptr, 0, 0,
                                              CM, CHID, b1m, nullptr, bng, bnb,
                                              nullptr, 0, h1, nullptr, CHID, 1);
    }

    // ---- MLP2 + residual (mode 3) ----
    {
        dim3 grid(CM / 32, CC / 64);
        gemm_kernel<<<grid, 256, 0, stream>>>(h1, CHID, W2h, CHID, CHID,
                                              nullptr, 0, nullptr, 0, 0,
                                              CM, CC, b2m, nullptr, nullptr, nullptr,
                                              x1, CC, nullptr, out, CC, 3);
    }
}
